// Set2Set_627065225778
// MI455X (gfx1250) — matrix operand order repaired and checked by reading, not compiled, not run on hardware
//
#include <hip/hip_runtime.h>
#include <hip/hip_bf16.h>

// ---------------------------------------------------------------------------
// Set2Set (B=256, N=256, D=512, H=1024) for gfx1250 (CDNA5, wave32, WMMA)
//
// Per step t = 0..255:
//   kernel 1: gates = X @ Wc^T  fused with LSTM cell -> c, q   (WMMA bf16,
//             async-to-LDS staging, gate columns co-located per block)
//   kernel 2: r = softmax(E_b q_b)^T E_b   (per-batch attention, bf16 E in L2)
// X double-buffered across steps. Final: relu([q;r] @ W_pred^T + b_pred).
// ---------------------------------------------------------------------------

typedef __attribute__((ext_vector_type(16))) __bf16 v16bf;
typedef __attribute__((ext_vector_type(8)))  __bf16 v8bf;
typedef __attribute__((ext_vector_type(8)))  float  v8f;

union ABFrag { v16bf v; v8bf h[2]; };

#define S2S_B 256
#define S2S_N 256
#define S2S_D 512
#define S2S_K 1024   // q_star width = 2*D
#define S2S_G 2048   // 4*D gate width

// ----------------------- CDNA5 async LDS staging ---------------------------
// global_load_async_to_lds_b128: per-lane 16B memory->LDS, tracked by ASYNCcnt.
__device__ __forceinline__ void async_copy_16B(const void* lds_dst, const void* gsrc) {
#if defined(__gfx1250__)
    unsigned lds_off = (unsigned)(unsigned long long)(uintptr_t)lds_dst; // addr[31:0] = LDS offset
    asm volatile("global_load_async_to_lds_b128 %0, %1, off"
                 :: "v"(lds_off), "v"((unsigned long long)(uintptr_t)gsrc)
                 : "memory");
#else
    (void)lds_dst; (void)gsrc;
#endif
}
__device__ __forceinline__ void wait_async() {
#if defined(__gfx1250__)
    asm volatile("s_wait_asynccnt 0" ::: "memory");
#endif
}

// -------------------------- weight / input prep ----------------------------

__global__ __launch_bounds__(256) void conv_emb_kernel(const float* __restrict__ src,
                                                       __bf16* __restrict__ dst, int n) {
    int i = blockIdx.x * 256 + threadIdx.x;
    if (i < n) dst[i] = (__bf16)src[i];
}

// Wc[n][k] = W_ih[n][k] + (k < 512 ? W_hh[n][k] : 0)   (exploits h == q)
__global__ __launch_bounds__(256) void build_wc_kernel(const float* __restrict__ W_ih,
                                                       const float* __restrict__ W_hh,
                                                       __bf16* __restrict__ Wc) {
    int i = blockIdx.x * 256 + threadIdx.x;          // over 2048*1024
    int n = i >> 10, k = i & 1023;
    float v = W_ih[(size_t)n * S2S_K + k];
    if (k < S2S_D) v += W_hh[(size_t)n * S2S_D + k];
    Wc[i] = (__bf16)v;
}

__global__ __launch_bounds__(256) void init_state_kernel(float* __restrict__ c,
                                                         __bf16* __restrict__ X0) {
    int i = blockIdx.x * 256 + threadIdx.x;          // over 262144
    X0[i] = (__bf16)0.0f;
    if (i < S2S_B * S2S_D) c[i] = 0.0f;
}

// --------------------- fused gates-GEMM + LSTM cell ------------------------
// Block tile: 64 batch rows x (32 d-columns x 4 gates). 8 waves (2x4),
// wave tile 32x32 -> wave (wave>>1) owns gate (wave>>1) for the whole d-tile.
// B row nl in [0,128): gate = nl>>5, dl = nl&31, Wc row = gate*512 + d0 + dl.

__global__ __launch_bounds__(256) void gemm_lstm_wmma(const __bf16* __restrict__ X,
                                                      const __bf16* __restrict__ Wc,
                                                      const float* __restrict__ b_ih,
                                                      const float* __restrict__ b_hh,
                                                      float* __restrict__ c,
                                                      float* __restrict__ q,
                                                      __bf16* __restrict__ Xout) {
    __shared__ __bf16 As[64][40];     // +8 pad: conflict-free fragment reads
    __shared__ __bf16 Bs[128][40];
    __shared__ float  Cs[64][132];    // 132-stride: lanes i / i+16 hit distinct banks

    const int tid  = threadIdx.x;
    const int lane = tid & 31;
    const int wave = tid >> 5;
    const int wm   = (wave & 1) * 32;
    const int wn   = (wave >> 1) * 32;
    const int m0   = blockIdx.y * 64;     // batch tile
    const int d0   = blockIdx.x * 32;     // d tile
    const int half = lane >> 4;
    const int l16  = lane & 15;

    v8f acc[2][2] = {};

    for (int k0 = 0; k0 < S2S_K; k0 += 32) {
        {   // A tile 64x32: one async 16B chunk per thread
            int row = tid >> 2, kc = (tid & 3) * 8;
            async_copy_16B(&As[row][kc], X + (size_t)(m0 + row) * S2S_K + k0 + kc);
        }
        #pragma unroll
        for (int j = 0; j < 2; ++j) {   // B tile 128x32: two async 16B chunks
            int cidx = j * 256 + tid;
            int nl = cidx >> 2, kc = (cidx & 3) * 8;
            int wrow = (nl >> 5) * S2S_D + d0 + (nl & 31);   // gate*512 + d
            async_copy_16B(&Bs[nl][kc], Wc + (size_t)wrow * S2S_K + k0 + kc);
        }
        wait_async();
        __syncthreads();

        ABFrag a[2], b[2];
        #pragma unroll
        for (int i = 0; i < 2; ++i) {
            const __bf16* p = &As[wm + i * 16 + l16][half * 8];
            a[i].h[0] = *(const v8bf*)p;
            a[i].h[1] = *(const v8bf*)(p + 16);
        }
        #pragma unroll
        for (int j = 0; j < 2; ++j) {
            const __bf16* p = &Bs[wn + j * 16 + l16][half * 8];
            b[j].h[0] = *(const v8bf*)p;
            b[j].h[1] = *(const v8bf*)(p + 16);
        }
        #pragma unroll
        for (int i = 0; i < 2; ++i)
            #pragma unroll
            for (int j = 0; j < 2; ++j)
                acc[i][j] = __builtin_amdgcn_wmma_f32_16x16x32_bf16(
                    false, a[i].v, false, b[j].v, (short)0, acc[i][j], false, false);
        __syncthreads();
    }

    // spill accumulators to LDS (C layout: lanes 0-15 M=r, lanes 16-31 M=r+8)
    #pragma unroll
    for (int i = 0; i < 2; ++i)
        #pragma unroll
        for (int j = 0; j < 2; ++j)
            #pragma unroll
            for (int r = 0; r < 8; ++r)
                Cs[wm + i * 16 + half * 8 + r][wn + j * 16 + l16] = acc[i][j][r];
    __syncthreads();

    // fused LSTM cell: 8 (b,d) pairs per thread over the 64x32 tile
    #pragma unroll
    for (int r = 0; r < 8; ++r) {
        int idx = r * 256 + tid;
        int ml = idx >> 5, dl = idx & 31;
        int bglob = m0 + ml, d = d0 + dl;
        float gi = Cs[ml][dl]       + b_ih[d]        + b_hh[d];
        float gf = Cs[ml][32 + dl]  + b_ih[d + 512]  + b_hh[d + 512];
        float gg = Cs[ml][64 + dl]  + b_ih[d + 1024] + b_hh[d + 1024];
        float go = Cs[ml][96 + dl]  + b_ih[d + 1536] + b_hh[d + 1536];
        float si = 1.0f / (1.0f + __expf(-gi));
        float sf = 1.0f / (1.0f + __expf(-gf));
        float so = 1.0f / (1.0f + __expf(-go));
        size_t ci = (size_t)bglob * S2S_D + d;
        float cn = sf * c[ci] + si * tanhf(gg);
        c[ci] = cn;
        float qq = so * tanhf(cn);
        q[ci] = qq;
        Xout[(size_t)bglob * S2S_K + d] = (__bf16)qq;   // q half of next q_star
    }
}

// ------------------------- generic WMMA GEMM (pred) ------------------------
// out[M x N] = relu( X[M x K] @ W[N x K]^T + bias )

__global__ __launch_bounds__(256) void gemm_bf16_wmma(const __bf16* __restrict__ X,
                                                      const __bf16* __restrict__ W,
                                                      const float* __restrict__ bias,
                                                      float* __restrict__ out,
                                                      int M, int N, int K) {
    __shared__ __bf16 As[64][40];
    __shared__ __bf16 Bs[128][40];

    const int tid  = threadIdx.x;
    const int lane = tid & 31;
    const int wave = tid >> 5;
    const int wm   = (wave & 1) * 32;
    const int wn   = (wave >> 1) * 32;
    const int m0   = blockIdx.y * 64;
    const int n0   = blockIdx.x * 128;
    const int half = lane >> 4;
    const int l16  = lane & 15;

    v8f acc[2][2] = {};

    for (int k0 = 0; k0 < K; k0 += 32) {
        {
            int row = tid >> 2, kc = (tid & 3) * 8;
            async_copy_16B(&As[row][kc], X + (size_t)(m0 + row) * K + k0 + kc);
        }
        #pragma unroll
        for (int j = 0; j < 2; ++j) {
            int cidx = j * 256 + tid;
            int n = cidx >> 2, kc = (cidx & 3) * 8;
            async_copy_16B(&Bs[n][kc], W + (size_t)(n0 + n) * K + k0 + kc);
        }
        wait_async();
        __syncthreads();

        ABFrag a[2], b[2];
        #pragma unroll
        for (int i = 0; i < 2; ++i) {
            const __bf16* p = &As[wm + i * 16 + l16][half * 8];
            a[i].h[0] = *(const v8bf*)p;
            a[i].h[1] = *(const v8bf*)(p + 16);
        }
        #pragma unroll
        for (int j = 0; j < 2; ++j) {
            const __bf16* p = &Bs[wn + j * 16 + l16][half * 8];
            b[j].h[0] = *(const v8bf*)p;
            b[j].h[1] = *(const v8bf*)(p + 16);
        }
        #pragma unroll
        for (int i = 0; i < 2; ++i)
            #pragma unroll
            for (int j = 0; j < 2; ++j)
                acc[i][j] = __builtin_amdgcn_wmma_f32_16x16x32_bf16(
                    false, a[i].v, false, b[j].v, (short)0, acc[i][j], false, false);
        __syncthreads();
    }

    #pragma unroll
    for (int i = 0; i < 2; ++i) {
        #pragma unroll
        for (int j = 0; j < 2; ++j) {
            int col = n0 + wn + j * 16 + l16;
            float bv = bias[col];
            #pragma unroll
            for (int r = 0; r < 8; ++r) {
                int row = m0 + wm + i * 16 + half * 8 + r;
                float v = fmaxf(acc[i][j][r] + bv, 0.0f);
                out[(size_t)row * N + col] = v;
            }
        }
    }
}

// ------------------------------ attention ----------------------------------
// One 256-thread block per batch. e=E q, softmax, r=a^T E. E bf16, L2-resident.

__global__ __launch_bounds__(256) void attention_kernel(const __bf16* __restrict__ emb,
                                                        const float* __restrict__ q,
                                                        __bf16* __restrict__ Xout) {
    const int b = blockIdx.x, tid = threadIdx.x;
    const int lane = tid & 31, wave = tid >> 5;
    __shared__ float qs[512];
    __shared__ float es[256];
    __shared__ float red[8];

    qs[tid]       = q[(size_t)b * S2S_D + tid];
    qs[tid + 256] = q[(size_t)b * S2S_D + tid + 256];
    __syncthreads();

    const __bf16* E = emb + (size_t)b * S2S_N * S2S_D;

    // phase 1: e[n] = <E[n,:], q>
    for (int it = 0; it < 32; ++it) {
        int n = it * 8 + wave;
        const __bf16* row = E + (size_t)n * S2S_D + lane * 16;
        v8bf x0 = *(const v8bf*)row;
        v8bf x1 = *(const v8bf*)(row + 8);
        float s = 0.0f;
        #pragma unroll
        for (int i = 0; i < 8; ++i)
            s += (float)x0[i] * qs[lane * 16 + i] +
                 (float)x1[i] * qs[lane * 16 + 8 + i];
        #pragma unroll
        for (int off = 16; off; off >>= 1) s += __shfl_xor(s, off, 32);
        if (lane == 0) es[n] = s;
    }
    __syncthreads();

    // phase 2: softmax over 256 scores
    float v = es[tid];
    float m = v;
    #pragma unroll
    for (int off = 16; off; off >>= 1) m = fmaxf(m, __shfl_xor(m, off, 32));
    if (lane == 0) red[wave] = m;
    __syncthreads();
    float mall = red[0];
    #pragma unroll
    for (int w = 1; w < 8; ++w) mall = fmaxf(mall, red[w]);
    float ex = __expf(v - mall);
    float s = ex;
    #pragma unroll
    for (int off = 16; off; off >>= 1) s += __shfl_xor(s, off, 32);
    __syncthreads();
    if (lane == 0) red[wave] = s;
    __syncthreads();
    float stot = 0.0f;
    #pragma unroll
    for (int w = 0; w < 8; ++w) stot += red[w];
    es[tid] = ex / stot;
    __syncthreads();

    // phase 3: r[d] = sum_n a[n] * E[n,d]; 2 d's per thread
    float a0 = 0.0f, a1 = 0.0f;
    for (int n = 0; n < 256; ++n) {
        if (n + 8 < 256)
            __builtin_prefetch(E + (size_t)(n + 8) * S2S_D + tid, 0, 1);
        float p = es[n];
        a0 += p * (float)E[(size_t)n * S2S_D + tid];
        a1 += p * (float)E[(size_t)n * S2S_D + tid + 256];
    }
    Xout[(size_t)b * S2S_K + S2S_D + tid]       = (__bf16)a0;  // r half
    Xout[(size_t)b * S2S_K + S2S_D + tid + 256] = (__bf16)a1;
}

// ------------------------------- launcher ----------------------------------

extern "C" void kernel_launch(void* const* d_in, const int* in_sizes, int n_in,
                              void* d_out, int out_size, void* d_ws, size_t ws_size,
                              hipStream_t stream) {
    const float* emb    = (const float*)d_in[0];   // [256,256,512]
    const float* W_ih   = (const float*)d_in[1];   // [2048,1024]
    const float* W_hh   = (const float*)d_in[2];   // [2048,512]
    const float* b_ih   = (const float*)d_in[3];   // [2048]
    const float* b_hh   = (const float*)d_in[4];   // [2048]
    const float* W_pred = (const float*)d_in[5];   // [512,1024]
    const float* b_pred = (const float*)d_in[6];   // [512]
    float* out = (float*)d_out;                    // [256,512]

    char* ws = (char*)d_ws;
    __bf16* embbf = (__bf16*)(ws);                          // 67,108,864 B
    __bf16* Wc    = (__bf16*)(ws + 67108864);               //  4,194,304 B
    __bf16* Wp    = (__bf16*)(ws + 71303168);               //  1,048,576 B
    __bf16* Xb[2];
    Xb[0]         = (__bf16*)(ws + 72351744);               //    524,288 B
    Xb[1]         = (__bf16*)(ws + 72876032);               //    524,288 B
    float*  c     = (float*) (ws + 73400320);               //    524,288 B
    float*  q     = (float*) (ws + 73924608);               //    524,288 B

    // one-time prep (inside graph; deterministic, inputs untouched)
    const int embN = S2S_B * S2S_N * S2S_D;                 // 33,554,432
    conv_emb_kernel<<<embN / 256, 256, 0, stream>>>(emb, embbf, embN);
    conv_emb_kernel<<<(S2S_D * S2S_K) / 256, 256, 0, stream>>>(W_pred, Wp, S2S_D * S2S_K);
    build_wc_kernel<<<(S2S_G * S2S_K) / 256, 256, 0, stream>>>(W_ih, W_hh, Wc);
    init_state_kernel<<<(S2S_B * S2S_K) / 256, 256, 0, stream>>>(c, Xb[0]);

    // recurrent steps: read X[t%2], write X[(t+1)%2]
    for (int t = 0; t < S2S_N; ++t) {
        const __bf16* Xin = Xb[t & 1];
        __bf16* Xout      = Xb[(t + 1) & 1];
        gemm_lstm_wmma<<<dim3(S2S_D / 32, S2S_B / 64), 256, 0, stream>>>(
            Xin, Wc, b_ih, b_hh, c, q, Xout);
        attention_kernel<<<S2S_B, 256, 0, stream>>>(embbf, q, Xout);
    }

    // prediction head: relu(q_star @ W_pred^T + b_pred); final X is Xb[0]
    gemm_bf16_wmma<<<dim3(S2S_D / 128, S2S_B / 64), 256, 0, stream>>>(
        Xb[0], Wp, b_pred, out, S2S_B, S2S_D, S2S_K);
    (void)in_sizes; (void)n_in; (void)out_size; (void)ws_size;
}
